// DiagonalLSTM_58918361367137
// MI455X (gfx1250) — matrix operand order repaired and checked by reading, not compiled, not run on hardware
//
#include <hip/hip_runtime.h>
#include <hip/hip_bf16.h>

// ---------------------------------------------------------------------------
// Diagonal LSTM on MI455X (gfx1250, wave32, WMMA).
//
//   B=16, C=64, H=64, W=64, HD=128, SW=H+W-1=127, G=5*HD=640
//
// Scan is sequential over 127 skewed columns but independent per batch.
// grid=16 workgroups (one per batch), 512 threads = 16 waves each.
// Per step: one fused GEMM  gates(640x64) = Wpack(640x320) @ [x;h;h_sh](320x64)
// using v_wmma_f32_16x16x32_f16, then an in-register LSTM cell update.
//
// Weight fragments are re-streamed from L2 every step (Apack is 400 KB and
// L2-resident); an opaque-OFFSET asm keeps LICM from hoisting the loads
// (which spilled 400 VGPRs to scratch in round 1) while preserving global
// address-space inference (round 2's opaque pointer demoted the loads to
// flat_load, which needlessly couples them to the DS counter).
// ---------------------------------------------------------------------------

typedef __attribute__((ext_vector_type(16))) _Float16 v16h;
typedef __attribute__((ext_vector_type(8)))  float    v8f;

#define NB   16
#define NC   64
#define NH   64
#define NW   64
#define NHD  128
#define NSW  127
#define NG   640           // 5*HD
#define KTOT 320           // C + 2*HD
#define NMT  40            // 640/16
#define NKT  10            // 320/32
#define NNT  4             // 64/16

// ws layout (bytes)
#define APACK_BYTES  (NG*KTOT*2)            // 409600
#define BIAS_OFF     APACK_BYTES
#define XPACK_OFF    (APACK_BYTES + 2560)   // 412160
#define XPACK_HALVES (NB*NSW*2*4*32*16)     // 8,323,072

__device__ __forceinline__ float fast_sigmoid(float x) {
    // 1 / (1 + exp2(-x*log2(e)))
    return __builtin_amdgcn_rcpf(1.0f + __builtin_amdgcn_exp2f(-x * 1.44269504088896f));
}
__device__ __forceinline__ float fast_tanh(float x) {
#if __has_builtin(__builtin_amdgcn_tanhf)
    return __builtin_amdgcn_tanhf(x);
#else
    // tanh(x) = 1 - 2/(exp(2x)+1),  exp(2x)=exp2(x*2*log2(e))
    float e = __builtin_amdgcn_exp2f(x * 2.88539008177793f);
    return 1.0f - 2.0f * __builtin_amdgcn_rcpf(e + 1.0f);
#endif
}
__device__ __forceinline__ float lane_gather(float v, int srcLane) {
    int iv = __builtin_bit_cast(int, v);
    int r  = __builtin_amdgcn_ds_bpermute(srcLane << 2, iv);
    return __builtin_bit_cast(float, r);
}

// ---------------------------------------------------------------------------
// Kernel 1: pack combined weights [W_is | W_ss1 | W_ss0] (640 x 320) into
// WMMA A-fragment order (16x32 f16 tiles, contiguous k order), and fuse biases.
// A tile layout: lane l (M = l&15), half q: K = 16*(l>>4) + q
// ---------------------------------------------------------------------------
__global__ void dlstm_pack_w(const float* __restrict__ w_is,
                             const float* __restrict__ w_ss,
                             const float* __restrict__ b_is,
                             const float* __restrict__ b_ss,
                             _Float16* __restrict__ Apack,
                             float* __restrict__ biasC) {
    int idx = blockIdx.x * 256 + threadIdx.x;
    if (idx < NG) biasC[idx] = b_is[idx] + b_ss[idx];
    if (idx >= NG * KTOT) return;
    int q    = idx & 15;
    int lane = (idx >> 4) & 31;
    int tile = idx >> 9;
    int kt   = tile % NKT;
    int mt   = tile / NKT;
    int m    = mt * 16 + (lane & 15);
    int kl   = ((lane >> 4) << 4) + q;
    int k    = kt * 32 + kl;
    float v;
    if (k < NC)            v = w_is[m * NC + k];                          // x  part
    else if (k < NC + NHD) v = w_ss[(m * NHD + (k - NC)) * 2 + 1];        // h_prev (w1)
    else                   v = w_ss[(m * NHD + (k - NC - NHD)) * 2 + 0];  // h_sh   (w0)
    Apack[idx] = (_Float16)v;
}

// ---------------------------------------------------------------------------
// Kernel 2: skew x and pack per-column x into dense-B fragment order.
// B tile (32K x 16N) layout: lane l (N = l&15), half q: K = 16*(l>>4) + q
// Out-of-band positions are zero (exactly reproduces skew()).
// ---------------------------------------------------------------------------
__global__ void dlstm_pack_x(const float* __restrict__ x,
                             _Float16* __restrict__ Xpack) {
    int idx = blockIdx.x * 256 + threadIdx.x;   // exactly XPACK_HALVES threads
    int q    = idx & 15;
    int lane = (idx >> 4) & 31;
    int nt   = (idx >> 9) & 3;
    int kt   = (idx >> 11) & 1;
    int rest = idx >> 12;
    int j    = rest % NSW;
    int b    = rest / NSW;
    int n = nt * 16 + (lane & 15);              // H row
    int c = kt * 32 + ((lane >> 4) << 4) + q;   // channel
    int w = j - n;
    float v = (w >= 0 && w < NW) ? x[(((b * NC + c) * NH) + n) * NW + w] : 0.0f;
    Xpack[idx] = (_Float16)v;
}

// ---------------------------------------------------------------------------
// Kernel 3: the scan.
// ---------------------------------------------------------------------------
__global__ __launch_bounds__(512)
void dlstm_scan(const _Float16* __restrict__ Apack,
                const float*    __restrict__ biasC,
                const _Float16* __restrict__ Xpack,
                float*          __restrict__ out) {
    const int tid  = threadIdx.x;
    const int lane = tid & 31;
    const int wv   = tid >> 5;      // 0..15
    const int dt   = wv & 7;        // HD tile (and gate m-tile offset)
    const int nth  = wv >> 3;       // n-tile half: {0,1}
    const int lo   = lane & 15;
    const int hi   = lane >> 4;
    const int b    = blockIdx.x;

    float* outB = out + (size_t)b * (NHD * NH * NW);

    // h feedback in B-fragment layout: [kt'=0..7][nt=0..3][lane=0..31][16 halves]
    __shared__ __attribute__((aligned(32))) _Float16 hfrag[8 * 4 * 32 * 16]; // 32 KB
    __shared__ float cbound[2][NHD];   // c[:,31] boundary between wave halves

    for (int i = tid; i < 8 * 4 * 32 * 16; i += 512) hfrag[i] = (_Float16)0.0f;
    if (tid < 2 * NHD) ((float*)cbound)[tid] = 0.0f;
    __syncthreads();

    // per-lane biases for the 5 gates this wave owns (persistent in VGPRs)
    float biasr[5][8];
#pragma unroll
    for (int g = 0; g < 5; ++g)
#pragma unroll
        for (int r = 0; r < 8; ++r)
            biasr[g][r] = biasC[g * NHD + dt * 16 + r + 8 * hi];

    // cell state c (this wave's d-range x its 32 H rows) lives in VGPRs
    float creg[2][8];
#pragma unroll
    for (int t = 0; t < 2; ++t)
#pragma unroll
        for (int r = 0; r < 8; ++r) creg[t][r] = 0.0f;

    const _Float16* Xb = Xpack + (size_t)b * NSW * 4096;

    for (int j = 0; j < NSW; ++j) {
        const _Float16* Xj = Xb + (size_t)j * 4096;
        if (j + 1 < NSW)
            __builtin_prefetch(Xj + 4096 + tid * 8, 0, 0);   // global_prefetch_b8

        // Opaque OFFSET: LICM can't hoist the 50 weight-fragment loads out of
        // the step loop (which spilled 400 VGPRs to scratch), but the base
        // pointer stays traceable to the global address space, so the loads
        // lower to global_load_b128 (LOADcnt only, decoupled from DS path).
        size_t aoff = 0;
        asm volatile("" : "+s"(aoff));
        const _Float16* Ap = Apack + aoff;

        v8f acc[5][2];
        const v8f vzero = {0.f, 0.f, 0.f, 0.f, 0.f, 0.f, 0.f, 0.f};
#pragma unroll
        for (int g = 0; g < 5; ++g)
#pragma unroll
            for (int t = 0; t < 2; ++t) acc[g][t] = vzero;

        // ---- fused GEMM: gates += Wpack(640x320) @ [x;h;h_sh](320x64) ----
#pragma unroll
        for (int kt = 0; kt < NKT; ++kt) {
            v16h bf[2];
#pragma unroll
            for (int t = 0; t < 2; ++t) {
                const int nt = nth * 2 + t;
                if (kt < 2)   // x part: fragments pre-packed in global (L2)
                    bf[t] = *reinterpret_cast<const v16h*>(Xj + ((kt * 4 + nt) * 32 + lane) * 16);
                else          // h / h_sh part: fragments in LDS
                    bf[t] = *reinterpret_cast<const v16h*>(&hfrag[(((kt - 2) * 4 + nt) * 32 + lane) * 16]);
            }
#pragma unroll
            for (int g = 0; g < 5; ++g) {
                const int mt = g * 8 + dt;
                v16h af = *reinterpret_cast<const v16h*>(Ap + (((mt * NKT + kt) * 32) + lane) * 16);
#pragma unroll
                for (int t = 0; t < 2; ++t)
                    acc[g][t] = __builtin_amdgcn_wmma_f32_16x16x32_f16(
                        false, af, false, bf[t], (short)0, acc[g][t], false, false);
            }
        }

        __syncthreads();   // all hfrag reads complete before rewrite

        const int p = j & 1;
        float cbv[8];
#pragma unroll
        for (int r = 0; r < 8; ++r) cbv[r] = cbound[p][dt * 16 + r + 8 * hi];

        float cold[2][8];
#pragma unroll
        for (int t = 0; t < 2; ++t)
#pragma unroll
            for (int r = 0; r < 8; ++r) cold[t][r] = creg[t][r];

        // ---- LSTM cell update, in registers ----
#pragma unroll
        for (int t = 0; t < 2; ++t) {
            const int nt = nth * 2 + t;
            const int n  = nt * 16 + lo;
#pragma unroll
            for (int r = 0; r < 8; ++r) {
                float o  = fast_sigmoid(acc[0][t][r] + biasr[0][r]);
                float fl = fast_sigmoid(acc[1][t][r] + biasr[1][r]);
                float fu = fast_sigmoid(acc[2][t][r] + biasr[2][r]);
                float ig = fast_sigmoid(acc[3][t][r] + biasr[3][r]);
                float gg = fast_tanh   (acc[4][t][r] + biasr[4][r]);

                float cp  = cold[t][r];
                // c shifted along H: lane-1 within 16-lane half-group
                float upv = lane_gather(cp, (lane & 16) | ((lane + 15) & 15));
                float pvt = lane_gather(cold[0][r], (lane & 16) | 15); // prev tile lane15
                float lb  = nth ? cbv[r] : 0.0f;                       // cross-wave / edge
                float csh = (lo != 0) ? upv : (t ? pvt : lb);

                float cn = fl * cp + fu * csh + ig * gg;
                creg[t][r] = cn;
                float h = o * fast_tanh(cn);

                const int d = dt * 16 + r + 8 * hi;
                const int w = j - n;
                if ((unsigned)w < (unsigned)NW)            // unskew: direct store
                    outB[(d * NH + n) * NW + w] = h;

                const _Float16 hh = (_Float16)h;
                // h_prev fragment slot: K = 64+d at row n
                const int k1 = NC + d;
                hfrag[(((k1 >> 5) - 2) * 4 + nt) * 512 + (lo + ((k1 >> 4) & 1) * 16) * 16 + (k1 & 15)] = hh;
                // h_sh fragment slot: K = 192+d at row n+1
                const int n2 = n + 1;
                if (n2 < NH) {
                    const int k2 = NC + NHD + d;
                    hfrag[(((k2 >> 5) - 2) * 4 + (n2 >> 4)) * 512 +
                          ((n2 & 15) + ((k2 >> 4) & 1) * 16) * 16 + (k2 & 15)] = hh;
                }
                if (n == 31) cbound[p ^ 1][d] = cn;   // boundary for next step
            }
        }
        __syncthreads();   // h fragments visible to next step's GEMM
    }
}

// ---------------------------------------------------------------------------
extern "C" void kernel_launch(void* const* d_in, const int* in_sizes, int n_in,
                              void* d_out, int out_size, void* d_ws, size_t ws_size,
                              hipStream_t stream) {
    const float* x    = (const float*)d_in[0];   // (16,64,64,64)
    const float* w_is = (const float*)d_in[1];   // (640,64)
    const float* b_is = (const float*)d_in[2];   // (640)
    const float* w_ss = (const float*)d_in[3];   // (640,128,2)
    const float* b_ss = (const float*)d_in[4];   // (640)
    float* out = (float*)d_out;                  // (16,128,64,64)

    char* ws = (char*)d_ws;
    _Float16* Apack = (_Float16*)ws;
    float*    biasC = (float*)(ws + BIAS_OFF);
    _Float16* Xpack = (_Float16*)(ws + XPACK_OFF);

    dlstm_pack_w<<<dim3((NG * KTOT + 255) / 256), dim3(256), 0, stream>>>(
        w_is, w_ss, b_is, b_ss, Apack, biasC);
    dlstm_pack_x<<<dim3(XPACK_HALVES / 256), dim3(256), 0, stream>>>(x, Xpack);
    dlstm_scan<<<dim3(NB), dim3(512), 0, stream>>>(Apack, biasC, Xpack, out);
}
